// PointTransformerLayer_1623497638700
// MI455X (gfx1250) — compile-verified, matrix operand order checked by reading, not run on hardware
//
#include <hip/hip_runtime.h>
#include <hip/hip_bf16.h>
#include <math.h>

typedef __attribute__((ext_vector_type(16))) _Float16 v16h;
typedef __attribute__((ext_vector_type(8)))  _Float16 v8h;
typedef __attribute__((ext_vector_type(8)))  float    v8f;

#define NPTS   8192
#define CH     256
#define NS     16
#define MIDS   32      // CH / SHARE
#define EPSBN  1e-5f

// ---------------------------------------------------------------------------
// WMMA fragment load: 16-bit A/B layout (ISA 7.12.2).
// lane L (row = L&15, hi = L>>4): halves at [k0 + hi*8 .. +8) and [k0+16+hi*8 .. +8)
// Works for both global and LDS (generic pointer).
// ---------------------------------------------------------------------------
__device__ inline v16h ldfrag(const _Float16* p) {
    v8h lo = *(const v8h*)(p);
    v8h hi = *(const v8h*)(p + 16);
    v16h f;
#pragma unroll
    for (int e = 0; e < 8; ++e) { f[e] = lo[e]; f[e + 8] = hi[e]; }
    return f;
}

// ---------------------------------------------------------------------------
// utility kernels
// ---------------------------------------------------------------------------
__global__ void cvt_f16_kernel(const float* __restrict__ s, _Float16* __restrict__ d, int n) {
    int i = blockIdx.x * blockDim.x + threadIdx.x;
    if (i < n) d[i] = (_Float16)s[i];
}

__global__ void pnorm_kernel(const float* __restrict__ p, float* __restrict__ p2, int n) {
    int i = blockIdx.x * blockDim.x + threadIdx.x;
    if (i < n) {
        float a = p[i * 3 + 0], b = p[i * 3 + 1], c = p[i * 3 + 2];
        p2[i] = a * a + b * b + c * c;
    }
}

// ---------------------------------------------------------------------------
// out[M,N] = A[M,K](f16) @ W[N,K](f16)^T + bias[N], f32 accum via WMMA.
// One wave per 16x16 output tile. K multiple of 32.
// ---------------------------------------------------------------------------
__global__ void wmma_gemm_xwT(const _Float16* __restrict__ A,
                              const _Float16* __restrict__ W,
                              const float* __restrict__ bias,
                              float* __restrict__ out,
                              int M, int N, int K) {
    const int lane = threadIdx.x & 31;
    const int wv   = threadIdx.x >> 5;
    const int tilesN = N >> 4;
    const int totalTiles = (M >> 4) * tilesN;
    const int t = blockIdx.x * (blockDim.x >> 5) + wv;   // wave-uniform
    if (t >= totalTiles) return;
    const int tm = t / tilesN, tn = t % tilesN;
    const int hi = lane >> 4, lr = lane & 15;

    const _Float16* arow = A + (size_t)(tm * 16 + lr) * K + hi * 8;
    const _Float16* brow = W + (size_t)(tn * 16 + lr) * K + hi * 8;

    v8f acc = {};
    for (int kk = 0; kk < K; kk += 32) {
        v16h a = ldfrag(arow + kk);
        v16h b = ldfrag(brow + kk);
        acc = __builtin_amdgcn_wmma_f32_16x16x32_f16(false, a, false, b,
                                                     (short)0, acc, false, false);
    }
    const float bc = bias[tn * 16 + lr];
    float* orow = out + (size_t)(tm * 16 + hi * 8) * N + tn * 16 + lr;
#pragma unroll
    for (int r = 0; r < 8; ++r) orow[(size_t)r * N] = acc[r] + bc;
}

// ---------------------------------------------------------------------------
// KNN: one 256-thread block per query point. d2 = p2[i]+p2[j]-2*p_i.p_j
// (same formula as reference). Per-thread sorted top-16 in registers
// (unrolled carry-swap insertion), then 16 rounds of block-wide argmin merge.
// ---------------------------------------------------------------------------
__global__ void knn_kernel(const float* __restrict__ p,
                           const float* __restrict__ p2,
                           int* __restrict__ knn) {
    const int i   = blockIdx.x;
    const int tid = threadIdx.x;

    const float pix = p[i * 3 + 0], piy = p[i * 3 + 1], piz = p[i * 3 + 2];
    const float p2i = p2[i];

    float bd[16]; int bi[16];
#pragma unroll
    for (int s = 0; s < 16; ++s) { bd[s] = 3.4e38f; bi[s] = -1; }

    for (int j = tid; j < NPTS; j += 256) {
        float dot = pix * p[j * 3 + 0] + piy * p[j * 3 + 1] + piz * p[j * 3 + 2];
        float cd  = p2i + p2[j] - 2.0f * dot;
        int   ci  = j;
        if (cd < bd[15]) {
#pragma unroll
            for (int s = 0; s < 16; ++s) {
                if (cd < bd[s]) {
                    float td = bd[s]; int ti = bi[s];
                    bd[s] = cd; bi[s] = ci;
                    cd = td; ci = ti;
                }
            }
        }
    }

    __shared__ float rd[256];
    __shared__ int   rid[256];
    __shared__ int   rt[256];
    __shared__ int   s_win;

    for (int r = 0; r < NS; ++r) {
        rd[tid] = bd[0]; rid[tid] = bi[0]; rt[tid] = tid;
        __syncthreads();
#pragma unroll
        for (int s = 128; s > 0; s >>= 1) {
            if (tid < s) {
                if (rd[tid + s] < rd[tid]) {
                    rd[tid] = rd[tid + s]; rid[tid] = rid[tid + s]; rt[tid] = rt[tid + s];
                }
            }
            __syncthreads();
        }
        if (tid == 0) { knn[i * NS + r] = rid[0]; s_win = rt[0]; }
        __syncthreads();
        if (tid == s_win) {   // pop my head (static unrolled shift)
#pragma unroll
            for (int s = 0; s < 15; ++s) { bd[s] = bd[s + 1]; bi[s] = bi[s + 1]; }
            bd[15] = 3.4e38f;
        }
        __syncthreads();
    }
}

// ---------------------------------------------------------------------------
// Fused per-point kernel: gather -> linear_p -> logits -> BN/ReLU ->
// WMMA(256->32) -> BN/ReLU -> WMMA(32->32) -> softmax(16) -> aggregation.
// 128 threads / block; waves 0,1 each own one 16-column WMMA tile (M=16=NS).
// ---------------------------------------------------------------------------
struct FusedParams {
    const float* p;
    const float* xq; const float* xk; const float* xv;
    const int*   knn;
    const _Float16* ww1h; const _Float16* ww2h;
    const float* Wp1; const float* bp1;
    const float* bnp_g; const float* bnp_b; const float* bnp_m; const float* bnp_v;
    const float* Wp2; const float* bp2;
    const float* bnw1_g; const float* bnw1_b; const float* bnw1_m; const float* bnw1_v;
    const float* bw1;
    const float* bnw2_g; const float* bnw2_b; const float* bnw2_m; const float* bnw2_v;
    const float* bw2;
    float* out;
};

__global__ void fused_point_kernel(FusedParams P) {
    const int i    = blockIdx.x;
    const int tid  = threadIdx.x;
    const int lane = tid & 31;
    const int wv   = tid >> 5;
    const int hi   = lane >> 4, lr = lane & 15;

    __shared__ __align__(16) float    s_pr[NS][CH];    // 16 KB, pr (f32) for aggregation
    __shared__ __align__(16) _Float16 s_w[NS][CH];     //  8 KB, WMMA1 A (post BN/ReLU logits)
    __shared__ __align__(16) _Float16 s_a2[NS][MIDS];  //  1 KB, WMMA2 A
    __shared__ float s_wl[NS][MIDS];                   //  2 KB, logits -> softmax weights
    __shared__ float s_t[NS][3];
    __shared__ int   s_idx[NS];

    if (tid < NS) s_idx[tid] = P.knn[i * NS + tid];
    __syncthreads();

    // ---- linear_p stage 1: Linear(3,3) + BN(3) + ReLU -------------------
    if (tid < NS * 3) {
        const int j = tid / 3, d = tid % 3;
        const int nj = s_idx[j];
        const float r0 = P.p[nj * 3 + 0] - P.p[i * 3 + 0];
        const float r1 = P.p[nj * 3 + 1] - P.p[i * 3 + 1];
        const float r2 = P.p[nj * 3 + 2] - P.p[i * 3 + 2];
        float v = P.Wp1[d * 3 + 0] * r0 + P.Wp1[d * 3 + 1] * r1 + P.Wp1[d * 3 + 2] * r2
                + P.bp1[d];
        const float sc = P.bnp_g[d] * rsqrtf(P.bnp_v[d] + EPSBN);
        v = (v - P.bnp_m[d]) * sc + P.bnp_b[d];
        s_t[j][d] = fmaxf(v, 0.0f);
    }
    __syncthreads();

    // ---- pr = Linear(3,256); logits = xk_g - xq + pr; BN(256)+ReLU ------
    for (int u = tid; u < NS * CH; u += 128) {
        const int j = u >> 8, c = u & (CH - 1);
        const float pr = P.Wp2[c * 3 + 0] * s_t[j][0]
                       + P.Wp2[c * 3 + 1] * s_t[j][1]
                       + P.Wp2[c * 3 + 2] * s_t[j][2] + P.bp2[c];
        s_pr[j][c] = pr;
        const int nj = s_idx[j];
        float lg = P.xk[(size_t)nj * CH + c] - P.xq[(size_t)i * CH + c] + pr;
        const float sc = P.bnw1_g[c] * rsqrtf(P.bnw1_v[c] + EPSBN);
        lg = (lg - P.bnw1_m[c]) * sc + P.bnw1_b[c];
        s_w[j][c] = (_Float16)fmaxf(lg, 0.0f);
    }
    __syncthreads();

    // ---- WMMA1: [16 x 256] @ Ww1[32,256]^T -> [16 x 32]; +b; BN(32)+ReLU -
    if (wv < 2) {
        const int tn = wv;
        const _Float16* ar = &s_w[0][0] + lr * CH + hi * 8;
        const _Float16* br = P.ww1h + (size_t)(tn * 16 + lr) * CH + hi * 8;
        v8f acc = {};
        for (int kk = 0; kk < CH; kk += 32) {
            v16h a = ldfrag(ar + kk);
            v16h b = ldfrag(br + kk);
            acc = __builtin_amdgcn_wmma_f32_16x16x32_f16(false, a, false, b,
                                                         (short)0, acc, false, false);
        }
        const int n = tn * 16 + lr;
        const float sc = P.bnw2_g[n] * rsqrtf(P.bnw2_v[n] + EPSBN);
#pragma unroll
        for (int r = 0; r < 8; ++r) {
            float v = acc[r] + P.bw1[n];
            v = (v - P.bnw2_m[n]) * sc + P.bnw2_b[n];
            s_a2[r + hi * 8][n] = (_Float16)fmaxf(v, 0.0f);
        }
    }
    __syncthreads();

    // ---- WMMA2: [16 x 32] @ Ww2[32,32]^T -> [16 x 32]; +b ----------------
    if (wv < 2) {
        const int tn = wv;
        v16h a = ldfrag(&s_a2[0][0] + lr * MIDS + hi * 8);
        v16h b = ldfrag(P.ww2h + (size_t)(tn * 16 + lr) * MIDS + hi * 8);
        v8f acc = {};
        acc = __builtin_amdgcn_wmma_f32_16x16x32_f16(false, a, false, b,
                                                     (short)0, acc, false, false);
        const int n = tn * 16 + lr;
#pragma unroll
        for (int r = 0; r < 8; ++r) s_wl[r + hi * 8][n] = acc[r] + P.bw2[n];
    }
    __syncthreads();

    // ---- softmax over the 16 neighbors, per channel ----------------------
    if (tid < MIDS) {
        const int c = tid;
        float m = -3.4e38f;
#pragma unroll
        for (int j = 0; j < NS; ++j) m = fmaxf(m, s_wl[j][c]);
        float sum = 0.0f;
#pragma unroll
        for (int j = 0; j < NS; ++j) { float e = __expf(s_wl[j][c] - m); s_wl[j][c] = e; sum += e; }
        const float inv = 1.0f / sum;
#pragma unroll
        for (int j = 0; j < NS; ++j) s_wl[j][c] *= inv;
    }
    __syncthreads();

    // ---- aggregation: out[u] = sum_j (xv_g + pr) * w[j][u % 32] ----------
    for (int u = tid; u < CH; u += 128) {
        const int cc = u & (MIDS - 1);
        float acc = 0.0f;
#pragma unroll
        for (int j = 0; j < NS; ++j) {
            const int nj = s_idx[j];
            acc += (P.xv[(size_t)nj * CH + u] + s_pr[j][u]) * s_wl[j][cc];
        }
        P.out[(size_t)i * CH + u] = acc;
    }
}

// ---------------------------------------------------------------------------
// host side
// ---------------------------------------------------------------------------
extern "C" void kernel_launch(void* const* d_in, const int* in_sizes, int n_in,
                              void* d_out, int out_size, void* d_ws, size_t ws_size,
                              hipStream_t stream) {
    // --- input mapping: detect flatten order by size signature ------------
    // insertion order: p,x,Wq,bq,Wk,bk,Wv,bv,Wp1,bp1,Wp2,bp2,Ww1,bw1,Ww2,bw2,
    //                  bnp(g,b,m,v), bnw1(g,b,m,v), bnw2(g,b,m,v)
    // sorted (jax tree_flatten): p,Wk,Wp1,Wp2,Wq,Wv,Ww1,Ww2,bk,bnp(b,g,m,v),
    //                  bnw1(b,g,m,v),bnw2(b,g,m,v),bp1,bp2,bq,bv,bw1,bw2,x
    const bool ins = (n_in > 1 && in_sizes[1] == NPTS * CH);
    const float *p, *x, *Wq, *bq, *Wk, *bk, *Wv, *bv, *Wp1, *bp1, *Wp2, *bp2;
    const float *Ww1, *bw1, *Ww2, *bw2;
    const float *bnp_g, *bnp_b, *bnp_m, *bnp_v;
    const float *bnw1_g, *bnw1_b, *bnw1_m, *bnw1_v;
    const float *bnw2_g, *bnw2_b, *bnw2_m, *bnw2_v;
    #define IN(k) ((const float*)d_in[(k)])
    if (ins) {
        p = IN(0);  x = IN(1);
        Wq = IN(2); bq = IN(3); Wk = IN(4); bk = IN(5); Wv = IN(6); bv = IN(7);
        Wp1 = IN(8); bp1 = IN(9); Wp2 = IN(10); bp2 = IN(11);
        Ww1 = IN(12); bw1 = IN(13); Ww2 = IN(14); bw2 = IN(15);
        bnp_g = IN(16);  bnp_b = IN(17);  bnp_m = IN(18);  bnp_v = IN(19);
        bnw1_g = IN(20); bnw1_b = IN(21); bnw1_m = IN(22); bnw1_v = IN(23);
        bnw2_g = IN(24); bnw2_b = IN(25); bnw2_m = IN(26); bnw2_v = IN(27);
    } else {
        p = IN(0);
        Wk = IN(1); Wp1 = IN(2); Wp2 = IN(3); Wq = IN(4); Wv = IN(5);
        Ww1 = IN(6); Ww2 = IN(7); bk = IN(8);
        bnp_b = IN(9);   bnp_g = IN(10);  bnp_m = IN(11);  bnp_v = IN(12);
        bnw1_b = IN(13); bnw1_g = IN(14); bnw1_m = IN(15); bnw1_v = IN(16);
        bnw2_b = IN(17); bnw2_g = IN(18); bnw2_m = IN(19); bnw2_v = IN(20);
        bp1 = IN(21); bp2 = IN(22); bq = IN(23); bv = IN(24);
        bw1 = IN(25); bw2 = IN(26); x = IN(27);
    }
    #undef IN

    // --- workspace carve-out (~30.5 MB) -----------------------------------
    size_t off = 0;
    auto carve = [&](size_t bytes) -> void* {
        void* q = (char*)d_ws + off;
        off += (bytes + 255) & ~(size_t)255;
        return q;
    };
    _Float16* xh   = (_Float16*)carve((size_t)NPTS * CH * 2);
    _Float16* wqh  = (_Float16*)carve((size_t)CH * CH * 2);
    _Float16* wkh  = (_Float16*)carve((size_t)CH * CH * 2);
    _Float16* wvh  = (_Float16*)carve((size_t)CH * CH * 2);
    _Float16* ww1h = (_Float16*)carve((size_t)MIDS * CH * 2);
    _Float16* ww2h = (_Float16*)carve((size_t)MIDS * MIDS * 2);
    float* xq = (float*)carve((size_t)NPTS * CH * 4);
    float* xk = (float*)carve((size_t)NPTS * CH * 4);
    float* xv = (float*)carve((size_t)NPTS * CH * 4);
    float* p2 = (float*)carve((size_t)NPTS * 4);
    int*   kn = (int*)carve((size_t)NPTS * NS * 4);
    (void)ws_size;

    // --- 1) conversions + norms ------------------------------------------
    auto cvt = [&](const float* s, _Float16* d, int n) {
        cvt_f16_kernel<<<(n + 255) / 256, 256, 0, stream>>>(s, d, n);
    };
    cvt(x,   xh,   NPTS * CH);
    cvt(Wq,  wqh,  CH * CH);
    cvt(Wk,  wkh,  CH * CH);
    cvt(Wv,  wvh,  CH * CH);
    cvt(Ww1, ww1h, MIDS * CH);
    cvt(Ww2, ww2h, MIDS * MIDS);
    pnorm_kernel<<<(NPTS + 255) / 256, 256, 0, stream>>>(p, p2, NPTS);

    // --- 2) q/k/v projections via WMMA -----------------------------------
    {
        const int tiles = (NPTS / 16) * (CH / 16);   // 8192
        const int blocks = tiles / 8;                // 8 waves / block
        wmma_gemm_xwT<<<blocks, 256, 0, stream>>>(xh, wqh, bq, xq, NPTS, CH, CH);
        wmma_gemm_xwT<<<blocks, 256, 0, stream>>>(xh, wkh, bk, xk, NPTS, CH, CH);
        wmma_gemm_xwT<<<blocks, 256, 0, stream>>>(xh, wvh, bv, xv, NPTS, CH, CH);
    }

    // --- 3) KNN -----------------------------------------------------------
    knn_kernel<<<NPTS, 256, 0, stream>>>(p, p2, kn);

    // --- 4) fused attention + aggregation --------------------------------
    FusedParams P;
    P.p = p; P.xq = xq; P.xk = xk; P.xv = xv; P.knn = kn;
    P.ww1h = ww1h; P.ww2h = ww2h;
    P.Wp1 = Wp1; P.bp1 = bp1;
    P.bnp_g = bnp_g; P.bnp_b = bnp_b; P.bnp_m = bnp_m; P.bnp_v = bnp_v;
    P.Wp2 = Wp2; P.bp2 = bp2;
    P.bnw1_g = bnw1_g; P.bnw1_b = bnw1_b; P.bnw1_m = bnw1_m; P.bnw1_v = bnw1_v;
    P.bw1 = bw1;
    P.bnw2_g = bnw2_g; P.bnw2_b = bnw2_b; P.bnw2_m = bnw2_m; P.bnw2_v = bnw2_v;
    P.bw2 = bw2;
    P.out = (float*)d_out;
    fused_point_kernel<<<NPTS, 128, 0, stream>>>(P);
}
